// MemoryOptimizedLiquidMambaModel_33440615367065
// MI455X (gfx1250) — compile-verified, matrix operand order checked by reading, not run on hardware
//
#include <hip/hip_runtime.h>
#include <hip/hip_bf16.h>
#include <math.h>

// ---------------- problem constants ----------------
#define B_ 16
#define L_ 1024
#define F_ 64
#define D_ 256
#define S_ 2
#define N_ 16
#define H_ 32
#define BL_ (B_ * L_)          // 16384 rows
#define C_ 16                  // scan chunks
#define T_ (L_ / C_)           // 64 steps per chunk
#define EPS_ 1e-5f

typedef __attribute__((ext_vector_type(16))) _Float16 v16h;
typedef __attribute__((ext_vector_type(8)))  _Float16 v8h;
typedef __attribute__((ext_vector_type(8)))  float    v8f;
typedef __attribute__((ext_vector_type(4)))  float    v4f;

// ---------------------------------------------------------------------------
// WMMA fragment helpers (gfx1250 wave32, V_WMMA_F32_16X16X32_F16)
// A (16x32 f16): lane l holds row m = l&15; khalf = l>>4.
//   v16h element i -> K = ks*32 + khalf*8 + (i<8 ? i : 16 + (i-8))
// B (32x16 f16): symmetric layout, n = l&15 in place of m.
// C/D (16x16 f32, 8 VGPRs): lane l -> n = l&15, vgpr r -> m = 8*(l>>4)+r.
// ---------------------------------------------------------------------------

__device__ __forceinline__ v16h load_a_frag_f16(const _Float16* __restrict__ row,
                                                int ks, int lane) {
  const int khalf = (lane >> 4) & 1;
  const _Float16* p = row + ks * 32 + khalf * 8;
  v8h lo = *(const v8h*)(p);
  v8h hi = *(const v8h*)(p + 16);
  v16h a;
#pragma unroll
  for (int i = 0; i < 8; ++i) { a[i] = lo[i]; a[8 + i] = hi[i]; }
  return a;
}

__device__ __forceinline__ v16h load_a_frag_f32(const float* __restrict__ row,
                                                int ks, int lane) {
  const int khalf = (lane >> 4) & 1;
  const float* p = row + ks * 32 + khalf * 8;
  v4f l0 = *(const v4f*)(p + 0);
  v4f l1 = *(const v4f*)(p + 4);
  v4f h0 = *(const v4f*)(p + 16);
  v4f h1 = *(const v4f*)(p + 20);
  v16h a;
#pragma unroll
  for (int i = 0; i < 4; ++i) {
    a[i]      = (_Float16)l0[i];
    a[4 + i]  = (_Float16)l1[i];
    a[8 + i]  = (_Float16)h0[i];
    a[12 + i] = (_Float16)h1[i];
  }
  return a;
}

// Packed B panel: fragment (nt, ks) occupies 512 consecutive halfs,
// lane-major, 16 halfs per lane (32B, vector-loadable).
__device__ __forceinline__ v16h load_b_frag(const _Float16* __restrict__ packed,
                                            int nt, int ks, int ksteps, int lane) {
  const _Float16* p = packed + (((size_t)(nt * ksteps + ks)) << 9) + lane * 16;
  return *(const v16h*)p;
}

__device__ __forceinline__ v8f wmma_f16(v16h a, v16h b, v8f c) {
  return __builtin_amdgcn_wmma_f32_16x16x32_f16(false, a, false, b,
                                                (short)0, c, false, false);
}

// ---------------------------------------------------------------------------
// Pack a KxN row-major f32 weight matrix into fragment-ordered f16 panels.
// ---------------------------------------------------------------------------
__global__ void k_pack(const float* __restrict__ W, _Float16* __restrict__ out,
                       int K, int N) {
  const int total = K * N;
  const int ksteps = K >> 5;
  for (int idx = blockIdx.x * blockDim.x + threadIdx.x; idx < total;
       idx += gridDim.x * blockDim.x) {
    const int i    = idx & 15;
    const int lane = (idx >> 4) & 31;
    const int frag = idx >> 9;
    const int ks   = frag % ksteps;
    const int nt   = frag / ksteps;
    const int khalf = lane >> 4;
    const int k = ks * 32 + khalf * 8 + ((i < 8) ? i : 16 + (i - 8));
    const int n = nt * 16 + (lane & 15);
    out[idx] = (_Float16)W[k * N + n];
  }
}

// ---------------------------------------------------------------------------
// H = x @ W_in + b_in ; write H in f32 (scan) and f16 (WMMA operand).
// grid (BL/16, D/64), block 32: each wave computes a 16x64 strip (4 n-tiles),
// reusing each A fragment across 4 WMMAs.  K=64 -> 2 k-steps.
// ---------------------------------------------------------------------------
__global__ __launch_bounds__(32) void k_gemm_in(
    const float* __restrict__ x, const _Float16* __restrict__ pWin,
    const float* __restrict__ b_in, float* __restrict__ H32,
    _Float16* __restrict__ H16) {
  const int lane   = threadIdx.x;
  const int mtile  = blockIdx.x;
  const int ntbase = blockIdx.y * 4;
  const int row    = mtile * 16 + (lane & 15);
  const float* rp  = x + (size_t)row * F_;
  v8f c[4] = {};
#pragma unroll
  for (int ks = 0; ks < 2; ++ks) {
    v16h a = load_a_frag_f32(rp, ks, lane);
#pragma unroll
    for (int j = 0; j < 4; ++j) {
      v16h b = load_b_frag(pWin, ntbase + j, ks, 2, lane);
      c[j] = wmma_f16(a, b, c[j]);
    }
  }
  const int mb = mtile * 16 + 8 * (lane >> 4);
#pragma unroll
  for (int j = 0; j < 4; ++j) {
    const int   n    = (ntbase + j) * 16 + (lane & 15);
    const float bias = b_in[n];
#pragma unroll
    for (int r = 0; r < 8; ++r) {
      float v = c[j][r] + bias;
      size_t off = (size_t)(mb + r) * D_ + n;
      H32[off] = v;
      H16[off] = (_Float16)v;
    }
  }
}

// ---------------------------------------------------------------------------
// delta[s] = softplus(H@Wd[s] + bd[s]) * sigmoid(H@Wtau[s])
// grid (BL/16, D/64, S), block 32.  16x64 strip, dual GEMM: 8 accumulators,
// each A fragment feeds 8 WMMAs.  K=256 -> 8 k-steps.
// ---------------------------------------------------------------------------
__global__ __launch_bounds__(32) void k_gemm_proj(
    const _Float16* __restrict__ H16, const _Float16* __restrict__ pWd,
    const _Float16* __restrict__ pWtau, const float* __restrict__ bd,
    float* __restrict__ delta) {
  const int lane   = threadIdx.x;
  const int mtile  = blockIdx.x;
  const int ntbase = blockIdx.y * 4;
  const int s      = blockIdx.z;
  const _Float16* pd  = pWd + (size_t)s * D_ * D_;
  const _Float16* pt  = pWtau + (size_t)s * D_ * D_;
  float* dout         = delta + (size_t)s * BL_ * D_;
  const int row       = mtile * 16 + (lane & 15);
  const _Float16* rp  = H16 + (size_t)row * D_;
  __builtin_prefetch(rp, 0, 1);  // global_prefetch_b8 on the A row
  v8f c1[4] = {}, c2[4] = {};
#pragma unroll
  for (int ks = 0; ks < 8; ++ks) {
    v16h a = load_a_frag_f16(rp, ks, lane);
#pragma unroll
    for (int j = 0; j < 4; ++j) {
      v16h bd_ = load_b_frag(pd, ntbase + j, ks, 8, lane);
      c1[j] = wmma_f16(a, bd_, c1[j]);
      v16h bt_ = load_b_frag(pt, ntbase + j, ks, 8, lane);
      c2[j] = wmma_f16(a, bt_, c2[j]);
    }
  }
  const int mb = mtile * 16 + 8 * (lane >> 4);
#pragma unroll
  for (int j = 0; j < 4; ++j) {
    const int   n  = (ntbase + j) * 16 + (lane & 15);
    const float bb = bd[s * D_ + n];
#pragma unroll
    for (int r = 0; r < 8; ++r) {
      float zd = c1[j][r] + bb;
      float sp = (zd > 20.f) ? zd : log1pf(expf(zd));     // softplus
      float sg = 1.f / (1.f + expf(-c2[j][r]));           // sigmoid
      dout[(size_t)(mb + r) * D_ + n] = sp * sg;
    }
  }
}

// ---------------------------------------------------------------------------
// Bm[s] = H@WB[s], Cm[s] = H@WC[s]   (N=16: single n-tile)
// grid (BL/16, S), block 32.
// ---------------------------------------------------------------------------
__global__ __launch_bounds__(32) void k_gemm_bc(
    const _Float16* __restrict__ H16, const _Float16* __restrict__ pWB,
    const _Float16* __restrict__ pWC, float* __restrict__ Bm,
    float* __restrict__ Cm) {
  const int lane  = threadIdx.x;
  const int mtile = blockIdx.x;
  const int s     = blockIdx.y;
  const _Float16* pb = pWB + (size_t)s * D_ * N_;
  const _Float16* pc = pWC + (size_t)s * D_ * N_;
  float* bmo = Bm + (size_t)s * BL_ * N_;
  float* cmo = Cm + (size_t)s * BL_ * N_;
  const int row      = mtile * 16 + (lane & 15);
  const _Float16* rp = H16 + (size_t)row * D_;
  v8f c1 = {}, c2 = {};
#pragma unroll
  for (int ks = 0; ks < 8; ++ks) {
    v16h a  = load_a_frag_f16(rp, ks, lane);
    v16h b1 = load_b_frag(pb, 0, ks, 8, lane);
    v16h b2 = load_b_frag(pc, 0, ks, 8, lane);
    c1 = wmma_f16(a, b1, c1);
    c2 = wmma_f16(a, b2, c2);
  }
  const int n  = lane & 15;
  const int mb = mtile * 16 + 8 * (lane >> 4);
#pragma unroll
  for (int r = 0; r < 8; ++r) {
    bmo[(size_t)(mb + r) * N_ + n] = c1[r];
    cmo[(size_t)(mb + r) * N_ + n] = c2[r];
  }
}

// ---------------------------------------------------------------------------
// Scan pass 1: chunked recurrence.  grid (B, S, C), block 256 (lane = d).
// Runs T_=64 sequential steps from a zero state, and accumulates the chunk's
// sum of deltas.  Since a_t = exp(delta_t * A), the cross-chunk decay is
// exp(A * sum(delta)) -- the combine pass applies it in closed form.
// ---------------------------------------------------------------------------
__global__ __launch_bounds__(256) void k_scan_chunk(
    const float* __restrict__ H32, const float* __restrict__ delta,
    const float* __restrict__ Bm, const float* __restrict__ A_log,
    float* __restrict__ chunkst, float* __restrict__ dsums) {
  const int d = threadIdx.x;
  const int b = blockIdx.x;
  const int s = blockIdx.y;
  const int c = blockIdx.z;
  const float* dbase = delta + (size_t)s * BL_ * D_;
  const float* bmb   = Bm + (size_t)s * BL_ * N_;
  float A[N_];
#pragma unroll
  for (int n = 0; n < N_; ++n)
    A[n] = -__expf(A_log[((size_t)s * D_ + d) * N_ + n]);
  float st[N_];
#pragma unroll
  for (int n = 0; n < N_; ++n) st[n] = 0.f;
  float dsum = 0.f;

  const int t0 = c * T_;
  for (int t = t0; t < t0 + T_; ++t) {
    const size_t idx = (size_t)b * L_ + t;
    const float hh = H32[idx * D_ + d];
    const float de = dbase[idx * D_ + d];
    dsum += de;
    const float dh = de * hh;
    const float* bm = bmb + idx * N_;   // block-uniform -> scalar loads
#pragma unroll
    for (int n = 0; n < N_; ++n) {
      const float dA = __expf(de * A[n]);   // de>=0, A<0 -> dA in (0,1]
      st[n] = dA * st[n] + dh * bm[n];
    }
  }
  const size_t cbase = (((size_t)(s * B_ + b) * C_ + c) * D_ + d);
  float* stout = chunkst + cbase * N_;
#pragma unroll
  for (int n = 0; n < N_; ++n) stout[n] = st[n];
  dsums[cbase] = dsum;
}

// ---------------------------------------------------------------------------
// Scan pass 2: combine chunk states with closed-form decay, then contract
// with Cm at t = L-1 and add the Dp skip.  grid (B, S), block 256.
// ---------------------------------------------------------------------------
__global__ __launch_bounds__(256) void k_scan_combine(
    const float* __restrict__ chunkst, const float* __restrict__ dsums,
    const float* __restrict__ H32, const float* __restrict__ Cm,
    const float* __restrict__ A_log, const float* __restrict__ Dp,
    float* __restrict__ ylast) {
  const int d = threadIdx.x;
  const int b = blockIdx.x;
  const int s = blockIdx.y;
  float A[N_];
#pragma unroll
  for (int n = 0; n < N_; ++n)
    A[n] = -__expf(A_log[((size_t)s * D_ + d) * N_ + n]);

  float fin[N_];
#pragma unroll
  for (int n = 0; n < N_; ++n) fin[n] = 0.f;

  const size_t base0 = (size_t)(s * B_ + b) * C_;
  float suf = 0.f;                       // suffix sum of deltas after chunk c
  for (int c = C_ - 1; c >= 0; --c) {
    const size_t cbase = (base0 + c) * D_ + d;
    const float* stc = chunkst + cbase * N_;
#pragma unroll
    for (int n = 0; n < N_; ++n)
      fin[n] += __expf(A[n] * suf) * stc[n];
    suf += dsums[cbase];
  }

  const size_t lidx = (size_t)b * L_ + (L_ - 1);
  const float* cm = Cm + (size_t)s * BL_ * N_ + lidx * N_;
  float y = 0.f;
#pragma unroll
  for (int n = 0; n < N_; ++n) y += fin[n] * cm[n];
  y += H32[lidx * D_ + d] * Dp[s * D_ + d];
  ylast[((size_t)s * B_ + b) * D_ + d] = y;
}

// ---------------------------------------------------------------------------
// Head: last = 0.5*(y0@Wout0 + y1@Wout1); LayerNorm; relu(z@W1+b1)@W2+b2.
// grid B, block 256.
// ---------------------------------------------------------------------------
__global__ __launch_bounds__(256) void k_head(
    const float* __restrict__ ylast, const float* __restrict__ Wout,
    const float* __restrict__ ln_g, const float* __restrict__ ln_b,
    const float* __restrict__ W1, const float* __restrict__ b1,
    const float* __restrict__ W2, const float* __restrict__ b2,
    float* __restrict__ out) {
  __shared__ float s_y0[D_], s_y1[D_], s_red[D_], s_z[D_], s_h1[H_];
  const int d = threadIdx.x;
  const int b = blockIdx.x;
  s_y0[d] = ylast[((size_t)0 * B_ + b) * D_ + d];
  s_y1[d] = ylast[((size_t)1 * B_ + b) * D_ + d];
  __syncthreads();

  const float* W0 = Wout;                    // stream 0: (D,D)
  const float* W1o = Wout + (size_t)D_ * D_; // stream 1
  float acc = 0.f;
  for (int e = 0; e < D_; ++e)
    acc += s_y0[e] * W0[(size_t)e * D_ + d] + s_y1[e] * W1o[(size_t)e * D_ + d];
  const float last = 0.5f * acc;

  // mean
  s_red[d] = last;
  __syncthreads();
  for (int off = 128; off > 0; off >>= 1) {
    if (d < off) s_red[d] += s_red[d + off];
    __syncthreads();
  }
  const float mu = s_red[0] * (1.f / D_);
  __syncthreads();
  // variance
  const float ctr = last - mu;
  s_red[d] = ctr * ctr;
  __syncthreads();
  for (int off = 128; off > 0; off >>= 1) {
    if (d < off) s_red[d] += s_red[d + off];
    __syncthreads();
  }
  const float var = s_red[0] * (1.f / D_);
  const float z = ctr * rsqrtf(var + EPS_) * ln_g[d] + ln_b[d];
  s_z[d] = z;
  __syncthreads();

  if (d < H_) {
    float a = b1[d];
    for (int e = 0; e < D_; ++e) a += s_z[e] * W1[(size_t)e * H_ + d];
    s_h1[d] = a > 0.f ? a : 0.f;
  }
  __syncthreads();
  if (d == 0) {
    float o = b2[0];
#pragma unroll
    for (int j = 0; j < H_; ++j) o += s_h1[j] * W2[j];
    out[b] = o;
  }
}

// ---------------------------------------------------------------------------
extern "C" void kernel_launch(void* const* d_in, const int* in_sizes, int n_in,
                              void* d_out, int out_size, void* d_ws,
                              size_t ws_size, hipStream_t stream) {
  (void)in_sizes; (void)n_in; (void)out_size; (void)ws_size;
  const float* x     = (const float*)d_in[0];
  const float* W_in  = (const float*)d_in[1];
  const float* b_in  = (const float*)d_in[2];
  const float* Wd    = (const float*)d_in[3];
  const float* bd    = (const float*)d_in[4];
  const float* WB    = (const float*)d_in[5];
  const float* WC    = (const float*)d_in[6];
  const float* Wtau  = (const float*)d_in[7];
  const float* A_log = (const float*)d_in[8];
  const float* Dp    = (const float*)d_in[9];
  const float* Wout  = (const float*)d_in[10];
  const float* ln_g  = (const float*)d_in[11];
  const float* ln_b  = (const float*)d_in[12];
  const float* W1    = (const float*)d_in[13];
  const float* b1    = (const float*)d_in[14];
  const float* W2    = (const float*)d_in[15];
  const float* b2    = (const float*)d_in[16];
  float* out = (float*)d_out;

  // ---- workspace layout ----
  char* w = (char*)d_ws;
  size_t off = 0;
  float*    H32     = (float*)(w + off);    off += (size_t)BL_ * D_ * 4;        // 16 MB
  _Float16* H16     = (_Float16*)(w + off); off += (size_t)BL_ * D_ * 2;        //  8 MB
  float*    delta   = (float*)(w + off);    off += (size_t)S_ * BL_ * D_ * 4;   // 32 MB
  float*    Bm      = (float*)(w + off);    off += (size_t)S_ * BL_ * N_ * 4;   //  2 MB
  float*    Cm      = (float*)(w + off);    off += (size_t)S_ * BL_ * N_ * 4;   //  2 MB
  float*    ylast   = (float*)(w + off);    off += (size_t)S_ * B_ * D_ * 4;    // 32 KB
  float*    chunkst = (float*)(w + off);    off += (size_t)S_ * B_ * C_ * D_ * N_ * 4; // 8 MB
  float*    dsums   = (float*)(w + off);    off += (size_t)S_ * B_ * C_ * D_ * 4;      // .5 MB
  _Float16* pWin    = (_Float16*)(w + off); off += (size_t)F_ * D_ * 2;
  _Float16* pWd     = (_Float16*)(w + off); off += (size_t)S_ * D_ * D_ * 2;
  _Float16* pWtau   = (_Float16*)(w + off); off += (size_t)S_ * D_ * D_ * 2;
  _Float16* pWB     = (_Float16*)(w + off); off += (size_t)S_ * D_ * N_ * 2;
  _Float16* pWC     = (_Float16*)(w + off); off += (size_t)S_ * D_ * N_ * 2;

  // ---- pack weights into WMMA-fragment-ordered f16 panels ----
  k_pack<<<64, 256, 0, stream>>>(W_in, pWin, F_, D_);
  for (int s = 0; s < S_; ++s) {
    k_pack<<<128, 256, 0, stream>>>(Wd   + (size_t)s * D_ * D_, pWd   + (size_t)s * D_ * D_, D_, D_);
    k_pack<<<128, 256, 0, stream>>>(Wtau + (size_t)s * D_ * D_, pWtau + (size_t)s * D_ * D_, D_, D_);
    k_pack<<<16, 256, 0, stream>>>(WB + (size_t)s * D_ * N_, pWB + (size_t)s * D_ * N_, D_, N_);
    k_pack<<<16, 256, 0, stream>>>(WC + (size_t)s * D_ * N_, pWC + (size_t)s * D_ * N_, D_, N_);
  }

  // ---- pipeline ----
  k_gemm_in     <<<dim3(BL_ / 16, D_ / 64),     32, 0, stream>>>(x, pWin, b_in, H32, H16);
  k_gemm_proj   <<<dim3(BL_ / 16, D_ / 64, S_), 32, 0, stream>>>(H16, pWd, pWtau, bd, delta);
  k_gemm_bc     <<<dim3(BL_ / 16, S_),          32, 0, stream>>>(H16, pWB, pWC, Bm, Cm);
  k_scan_chunk  <<<dim3(B_, S_, C_),           256, 0, stream>>>(H32, delta, Bm, A_log, chunkst, dsums);
  k_scan_combine<<<dim3(B_, S_),               256, 0, stream>>>(chunkst, dsums, H32, Cm, A_log, Dp, ylast);
  k_head        <<<B_,                         256, 0, stream>>>(ylast, Wout, ln_g, ln_b, W1, b1, W2, b2, out);
}
